// BitLinearPacked_30124900614508
// MI455X (gfx1250) — matrix operand order repaired and checked by reading, not compiled, not run on hardware
//
#include <hip/hip_runtime.h>
#include <hip/hip_bf16.h>

typedef __attribute__((ext_vector_type(16))) _Float16 v16h;
typedef __attribute__((ext_vector_type(8)))  _Float16 v8h;
typedef __attribute__((ext_vector_type(8)))  float    v8f;
typedef __attribute__((ext_vector_type(4)))  float    v4f;

#define IN_DIM  1024
#define OUT_DIM 1024

// ---------------------------------------------------------------------------
// Kernel 1: unpack sign bits -> f16 (+1/-1) weight matrix Wh[n][k], row-major.
// packed is int32, one byte of bits per element, MSB-first (shift 7..0),
// element (n, c) covers w[n][8c .. 8c+7].  Wh offset = n*1024 + 8c = 8*idx.
// ---------------------------------------------------------------------------
__global__ __launch_bounds__(256) void bitlinear_unpack_kernel(
    const int* __restrict__ packed, _Float16* __restrict__ wh) {
  int idx = blockIdx.x * 256 + threadIdx.x;   // 0 .. OUT*IN/8-1
  int v = packed[idx];
  v8h h;
#pragma unroll
  for (int j = 0; j < 8; ++j) {
    h[j] = (_Float16)((((v >> (7 - j)) & 1) << 1) - 1);  // bit -> +/-1
  }
  *(v8h*)(wh + (size_t)idx * 8) = h;  // 16-byte store
}

// ---------------------------------------------------------------------------
// Kernel 2: y[M,N] = x[M,K] @ Wh[N,K]^T  via v_wmma_f32_16x16x32_f16.
// Wave tile: 32(M) x 64(N). Block: 8 waves -> 64(M) x 256(N).
// A fragment (16x32 f16): lane m=lane%16, halves 0..15 = K 16*lh+[0..15],
//                         (lh = lane>>4), same k order as B
// B fragment (32x16 f16): lane n=lane%16, halves 0..15 = K 16*lh+[0..15]
//   -> contiguous 32 bytes of Wh row n (Wh is w's native [N][K] layout).
// C/D: vgpr v <-> (m = v + 8*lh, n = lane%16).
// ---------------------------------------------------------------------------
__global__ __launch_bounds__(256) void bitlinear_gemm_kernel(
    const float* __restrict__ x, const _Float16* __restrict__ wh,
    float* __restrict__ out) {
  const int lane  = threadIdx.x & 31;
  const int wid   = threadIdx.x >> 5;   // 0..7
  const int lmod  = lane & 15;
  const int lh    = lane >> 4;          // 0 or 1
  const int waveM = wid & 1;            // 2 waves along M
  const int waveN = wid >> 1;           // 4 waves along N

  const int m0 = blockIdx.x * 64  + waveM * 32;   // wave M base (2 subtiles)
  const int n0 = blockIdx.y * 256 + waveN * 64;   // wave N base (4 subtiles)

  // Per-lane base pointers.
  const float* xrow0 = x + (size_t)(m0 + lmod) * IN_DIM + 16 * lh;
  const float* xrow1 = xrow0 + (size_t)16 * IN_DIM;
  const _Float16* wrow[4];
#pragma unroll
  for (int j = 0; j < 4; ++j)
    wrow[j] = wh + (size_t)(n0 + j * 16 + lmod) * IN_DIM + 16 * lh;

  v8f acc[2][4] = {};

  for (int k = 0; k < IN_DIM; k += 32) {
    // ---- A fragments: load fp32, convert to f16 in-register ----
    v16h a[2];
#pragma unroll
    for (int i = 0; i < 2; ++i) {
      const float* p = (i == 0 ? xrow0 : xrow1) + k;
      v4f f0 = *(const v4f*)(p);
      v4f f1 = *(const v4f*)(p + 4);
      v4f f2 = *(const v4f*)(p + 8);
      v4f f3 = *(const v4f*)(p + 12);
#pragma unroll
      for (int t = 0; t < 4; ++t) {
        a[i][t]      = (_Float16)f0[t];
        a[i][4 + t]  = (_Float16)f1[t];
        a[i][8 + t]  = (_Float16)f2[t];
        a[i][12 + t] = (_Float16)f3[t];
      }
    }

    // ---- B fragments: 32 contiguous bytes per lane (2x b128) ----
    v16h b[4];
#pragma unroll
    for (int j = 0; j < 4; ++j)
      b[j] = *(const v16h*)(wrow[j] + k);

    // ---- 8 WMMAs ----
#pragma unroll
    for (int i = 0; i < 2; ++i)
#pragma unroll
      for (int j = 0; j < 4; ++j)
        acc[i][j] = __builtin_amdgcn_wmma_f32_16x16x32_f16(
            /*neg_a=*/false, a[i], /*neg_b=*/false, b[j],
            /*c_mod=*/(short)0, acc[i][j],
            /*reuse_a=*/false, /*reuse_b=*/false);
  }

  // ---- Epilogue: C layout -> coalesced dword stores ----
#pragma unroll
  for (int i = 0; i < 2; ++i) {
    const int mbase = m0 + i * 16 + 8 * lh;
#pragma unroll
    for (int j = 0; j < 4; ++j) {
      const int n = n0 + j * 16 + lmod;
      float* o = out + (size_t)mbase * OUT_DIM + n;
#pragma unroll
      for (int v = 0; v < 8; ++v)
        o[(size_t)v * OUT_DIM] = acc[i][j][v];
    }
  }
}

// ---------------------------------------------------------------------------
extern "C" void kernel_launch(void* const* d_in, const int* in_sizes, int n_in,
                              void* d_out, int out_size, void* d_ws, size_t ws_size,
                              hipStream_t stream) {
  const float* x      = (const float*)d_in[0];  // [2,8192,1024] fp32
  const int*   packed = (const int*)d_in[1];    // [1024,128] int32 (8 bits each)
  float*       out    = (float*)d_out;          // [2,8192,1024] fp32
  _Float16*    wh     = (_Float16*)d_ws;        // 1024*1024 f16 = 2 MB scratch

  const int M = in_sizes[0] / IN_DIM;           // 16384

  // 1) unpack packed sign bits -> f16 +/-1 matrix (2 MB, L2-resident)
  const int n_packed = OUT_DIM * (IN_DIM / 8);  // 131072
  bitlinear_unpack_kernel<<<n_packed / 256, 256, 0, stream>>>(packed, wh);

  // 2) WMMA GEMM: block tile 64(M) x 256(N)
  dim3 grid(M / 64, OUT_DIM / 256);
  bitlinear_gemm_kernel<<<grid, 256, 0, stream>>>(x, wh, out);
}
